// ModulatedDeformConv_41188736369001
// MI455X (gfx1250) — matrix operand order repaired and checked by reading, not compiled, not run on hardware
//
#include <hip/hip_runtime.h>
#include <hip/hip_bf16.h>

// ---------------------------------------------------------------------------
// DCNv2 forward, fused im2col (bilinear, modulated) + bf16 WMMA GEMM.
// Fixed problem: N=8, Cin=Cout=256, H=W=Ho=Wo=64, KH=KW=3, stride=1, pad=1.
// GEMM: M=Cout=256, P=N*Ho*Wo=32768, Kdim=Cin*9=2304 (K-major: kk = k*256+c).
// Single M-block covers all 256 cout rows -> each im2col element is gathered
// exactly once (gathers are the dominant cost at ~300M 4B L2 hits).
// ---------------------------------------------------------------------------

typedef __attribute__((ext_vector_type(16))) __bf16 v16bf;
typedef __attribute__((ext_vector_type(8)))  float  v8f;

#define NB    8
#define CIN   256
#define COUT  256
#define HH    64
#define WW    64
#define HW    (HH * WW)          // 4096
#define KTAP  9
#define KDIM  (CIN * KTAP)       // 2304
#define PTOT  (NB * HW)          // 32768

#define MT    256                // cout tile == full Cout (no redundant gathers)
#define PT    64                 // pixel tile
#define KC    32                 // K chunk (bf16 WMMA depth)
#define NITER (KDIM / KC)        // 72

// workspace layout (bytes)
#define WT_OFF   0ull
#define WT_BYTES ((size_t)COUT * KDIM * 2)                 // 1,179,648
#define WG_OFF   (WT_OFF + WT_BYTES)                       // float4 taps
#define WG_BYTES ((size_t)NB * KTAP * HW * 16)             // 4,718,592
#define IX_OFF   (WG_OFF + WG_BYTES)                       // int4 indices
#define IX_BYTES ((size_t)NB * KTAP * HW * 16)

__device__ __forceinline__ unsigned short f2bf(float v) {
  unsigned u = __float_as_uint(v);
  u = (u + 0x7FFFu + ((u >> 16) & 1u)) >> 16;
  return (unsigned short)u;
}

__device__ __forceinline__ unsigned pack2bf(float a, float b) {
  return (unsigned)f2bf(a) | ((unsigned)f2bf(b) << 16);
}

// ---------------------------------------------------------------------------
// Kernel 1: weight [Cout][Cin][3][3] f32 -> Wt bf16 [Cout][kk], kk = k*256+c
// ---------------------------------------------------------------------------
__global__ void dcn_pack_weight(const float* __restrict__ w,
                                unsigned short* __restrict__ wt) {
  int i = blockIdx.x * blockDim.x + threadIdx.x;      // [0, COUT*KDIM)
  if (i >= COUT * KDIM) return;
  int cout = i / KDIM;
  int kk   = i - cout * KDIM;
  int k    = kk >> 8;            // /256
  int c    = kk & 255;
  wt[i] = f2bf(w[(cout * CIN + c) * KTAP + k]);
}

// ---------------------------------------------------------------------------
// Kernel 2: per (n,k,ho,wo) precompute 4 bilinear tap weights (mask-modulated,
// zeroed outside image) and 4 clamped flat indices (y*W+x).
// ---------------------------------------------------------------------------
__global__ void dcn_prep_sampling(const float* __restrict__ off,
                                  const float* __restrict__ msk,
                                  float4* __restrict__ wgt4,
                                  int4*   __restrict__ idx4) {
  int s = blockIdx.x * blockDim.x + threadIdx.x;      // [0, NB*KTAP*HW)
  if (s >= NB * KTAP * HW) return;
  int n  = s / (KTAP * HW);
  int r  = s - n * (KTAP * HW);
  int k  = r / HW;
  int hw = r - k * HW;
  int ho = hw >> 6;
  int wo = hw & 63;

  float dy = off[((n * (2 * KTAP)) + 2 * k + 0) * HW + hw];
  float dx = off[((n * (2 * KTAP)) + 2 * k + 1) * HW + hw];
  float m  = msk[((n * KTAP) + k) * HW + hw];

  float y  = (float)(ho - 1 + (k / 3)) + dy;
  float x  = (float)(wo - 1 + (k % 3)) + dx;
  float y0 = floorf(y), x0 = floorf(x);

  float4 wv; int4 iv;
  float* wp = &wv.x; int* ip = &iv.x;
#pragma unroll
  for (int t = 0; t < 4; ++t) {
    float yc = y0 + (float)(t >> 1);
    float xc = x0 + (float)(t & 1);
    float wgt = (1.0f - fabsf(y - yc)) * (1.0f - fabsf(x - xc)) * m;
    bool valid = (yc >= 0.0f) & (yc < (float)HH) & (xc >= 0.0f) & (xc < (float)WW);
    wp[t] = valid ? wgt : 0.0f;
    int yi = (int)fminf(fmaxf(yc, 0.0f), (float)(HH - 1));
    int xi = (int)fminf(fmaxf(xc, 0.0f), (float)(WW - 1));
    ip[t] = yi * WW + xi;
  }
  wgt4[s] = wv;
  idx4[s] = iv;
}

// ---------------------------------------------------------------------------
// Fragment loader: 16x32 bf16 fragment from an LDS tile stored row-major as
// u32 pairs (row stride = 16 u32). Layout per ISA 7.12.2 (16-bit A 16x32):
// lane half selects K half; VGPR j holds K pair (j<4 ? half*8+2j : 16+half*8+..).
// ---------------------------------------------------------------------------
__device__ __forceinline__ v16bf load_frag(const unsigned* __restrict__ lds,
                                           int rowBase, int lane) {
  struct U { unsigned u[8]; } t;
  int half = (lane >> 4) & 1;
  int row  = rowBase + (lane & 15);
  const unsigned* p = lds + row * 16;
#pragma unroll
  for (int j = 0; j < 8; ++j) {
    int kp = (j < 4) ? (half * 4 + j) : (8 + half * 4 + (j - 4));
    t.u[j] = p[kp];
  }
  return __builtin_bit_cast(v16bf, t);
}

// ---------------------------------------------------------------------------
// Kernel 3: fused im2col + GEMM.
// Block: 256 threads (8 waves tiling M). Tile: 256 cout x 64 px.
// Each wave: 32 cout x 64 px = 2x4 WMMA fragments per K chunk.
// ---------------------------------------------------------------------------
__global__ void __launch_bounds__(256)
dcn_gemm_wmma(const float* __restrict__ x,
              const unsigned* __restrict__ wtU,     // Wt as u32 pairs
              const float4* __restrict__ wgt4,
              const int4*   __restrict__ idx4,
              const float*  __restrict__ bias,
              float* __restrict__ out) {
  __shared__ unsigned Alds[MT * (KC / 2)];   // [m][kpair]  256*16 u32 = 16KB
  __shared__ unsigned Blds[PT * (KC / 2)];   // [p][kpair]   64*16 u32 =  4KB

  const int tid  = threadIdx.x;
  const int lane = tid & 31;
  const int wave = tid >> 5;                 // 0..7 along M

  const int p0 = blockIdx.x * PT;            // pixel base

  // B-staging role: thread -> one pixel column, 8 consecutive channels
  const int p_local = tid & 63;
  const int c_grp   = tid >> 6;              // 0..3
  const int pix     = p0 + p_local;
  const int n_img   = pix >> 12;             // /4096
  const int hw      = pix & 4095;
  const float* xn   = x + (size_t)n_img * CIN * HW;

  v8f acc[2][4];
#pragma unroll
  for (int a = 0; a < 2; ++a)
#pragma unroll
    for (int b = 0; b < 4; ++b)
      acc[a][b] = (v8f){0.f, 0.f, 0.f, 0.f, 0.f, 0.f, 0.f, 0.f};

  for (int q = 0; q < NITER; ++q) {
    const int k  = q >> 3;                   // kernel tap for this chunk
    const int c0 = (q & 7) * KC;             // channel base of chunk

    // ---- stage B tile: bilinear-sample 8 channels for this pixel ----
    // Tap weights/indices depend only on (n,k,ho,wo): one float4+int4 load,
    // reused across all 8 channels (and all cout, since MT == Cout).
    const int s = (n_img * KTAP + k) * HW + hw;
    const float4 wv = wgt4[s];
    const int4   iv = idx4[s];
    const int cbase = c0 + c_grp * 8;
#pragma unroll
    for (int jj = 0; jj < 4; ++jj) {
      const float* pc0 = xn + (size_t)(cbase + 2 * jj) * HW;
      const float* pc1 = pc0 + HW;
      float v0 = wv.x * pc0[iv.x] + wv.y * pc0[iv.y] +
                 wv.z * pc0[iv.z] + wv.w * pc0[iv.w];
      float v1 = wv.x * pc1[iv.x] + wv.y * pc1[iv.y] +
                 wv.z * pc1[iv.z] + wv.w * pc1[iv.w];
      Blds[p_local * 16 + (c_grp * 4) + jj] = pack2bf(v0, v1);
    }

    // ---- stage A tile: 16 u32 (32 bf16 weights) per thread ----
#pragma unroll
    for (int j = 0; j < 16; ++j) {
      int u  = tid * 16 + j;                 // [0, 4096)
      int m  = u >> 4;
      int kp = u & 15;
      Alds[u] = wtU[(size_t)m * (KDIM / 2) + q * (KC / 2) + kp];
    }

    // prefetch next chunk's weights into cache (global_prefetch_b8)
    if (q + 1 < NITER)
      __builtin_prefetch((const char*)(wtU + (size_t)(tid) * (KDIM / 2)
                                       + (q + 1) * (KC / 2)), 0, 1);

    __syncthreads();

    // ---- fragments + WMMA: wave owns rows [wave*32, wave*32+32) ----
    v16bf bfr[4];
#pragma unroll
    for (int fp = 0; fp < 4; ++fp)
      bfr[fp] = load_frag(Blds, fp * 16, lane);

#pragma unroll
    for (int fm = 0; fm < 2; ++fm) {
      v16bf afr = load_frag(Alds, wave * 32 + fm * 16, lane);
#pragma unroll
      for (int fp = 0; fp < 4; ++fp)
        acc[fm][fp] = __builtin_amdgcn_wmma_f32_16x16x32_bf16(
            false, afr, false, bfr[fp], (short)0, acc[fm][fp],
            false, false);
    }

    __syncthreads();
  }

  // ---- epilogue: D layout (ISA 7.12.2): lane half selects M+8, N = lane&15 ----
  const int nc    = lane & 15;
  const int mhalf = (lane >> 4) & 1;
#pragma unroll
  for (int fm = 0; fm < 2; ++fm) {
#pragma unroll
    for (int fp = 0; fp < 4; ++fp) {
      int pp  = p0 + fp * 16 + nc;
      int n2  = pp >> 12;
      int hw2 = pp & 4095;
#pragma unroll
      for (int r = 0; r < 8; ++r) {
        int cout = wave * 32 + fm * 16 + r + 8 * mhalf;
        out[((size_t)n2 * COUT + cout) * HW + hw2] = acc[fm][fp][r] + bias[cout];
      }
    }
  }
}

// ---------------------------------------------------------------------------
extern "C" void kernel_launch(void* const* d_in, const int* in_sizes, int n_in,
                              void* d_out, int out_size, void* d_ws, size_t ws_size,
                              hipStream_t stream) {
  const float* x      = (const float*)d_in[0];
  const float* offset = (const float*)d_in[1];
  const float* mask   = (const float*)d_in[2];
  const float* weight = (const float*)d_in[3];
  const float* bias   = (const float*)d_in[4];
  float* out          = (float*)d_out;

  char* ws = (char*)d_ws;
  unsigned short* wt = (unsigned short*)(ws + WT_OFF);
  float4* wgt4       = (float4*)(ws + WG_OFF);
  int4*   idx4       = (int4*)(ws + IX_OFF);

  (void)in_sizes; (void)n_in; (void)out_size; (void)ws_size;

  // 1) pack weights to bf16, K-major (kk = k*256 + c)
  {
    int total = COUT * KDIM;
    dcn_pack_weight<<<(total + 255) / 256, 256, 0, stream>>>(weight, wt);
  }
  // 2) precompute bilinear taps
  {
    int total = NB * KTAP * HW;
    dcn_prep_sampling<<<(total + 255) / 256, 256, 0, stream>>>(offset, mask,
                                                               wgt4, idx4);
  }
  // 3) fused im2col + WMMA GEMM (one block covers all 256 cout)
  {
    dim3 grid(PTOT / PT, 1);   // (512, 1)
    dcn_gemm_wmma<<<grid, 256, 0, stream>>>(x, (const unsigned*)wt,
                                            wgt4, idx4, bias, out);
  }
}
